// LatencyPredictor_47931835023811
// MI455X (gfx1250) — matrix operand order repaired and checked by reading, not compiled, not run on hardware
//
#include <hip/hip_runtime.h>

// ---------------------------------------------------------------------------
// CDNA5 (gfx1250) dendrite + LIF latency predictor.
//  - H[16x64] x W1^T[64x16] on v_wmma_f32_16x16x32_f16 (f32 accumulate)
//  - 4 tiles per wave: B-matrix (W1) fragments packed once, reused 4x
//  - Hankel slice staged in LDS (576 floats/WG), clamped at the tail
//  - LIF scan: 8x unrolled load batching + global_prefetch_b8 lookahead
// ---------------------------------------------------------------------------

typedef __attribute__((ext_vector_type(16))) _Float16 v16h;
typedef __attribute__((ext_vector_type(8)))  float    v8f;

#define Bsz    256
#define Lfull  4096
#define Tw     64
#define NK1    16
#define NK2    8
#define LP     (Lfull - Tw + 1)      /* 4033 */
#define ROWS_PER_WG 512              /* 8 waves x 4 tiles x 16 rows */
#define TILES_PER_WAVE 4
#define TAUf   20.0f
#define THf    1.0f

/* output layout (floats, per branch): d2 | idx | vals | lat_hard | lat_soft */
#define SZD       ((size_t)Bsz * (size_t)LP * NK2)
#define SZI       ((size_t)Bsz * (size_t)LP)
#define BR_STRIDE (SZD + 2*SZI + 2*(size_t)Bsz)

__global__ __launch_bounds__(256) void dendrite_wmma_kernel(
    const float* __restrict__ xa,  const float* __restrict__ xb,
    const float* __restrict__ W1a, const float* __restrict__ b1a,
    const float* __restrict__ W2a, const float* __restrict__ b2a,
    const float* __restrict__ W1b, const float* __restrict__ b1b,
    const float* __restrict__ W2b, const float* __restrict__ b2b,
    float* __restrict__ out, float* __restrict__ IT, int useT)
{
    const int branch = blockIdx.z;
    const int b      = blockIdx.y;
    const int l0g    = blockIdx.x * ROWS_PER_WG;

    const float* x  = branch ? xb  : xa;
    const float* W1 = branch ? W1b : W1a;
    const float* b1 = branch ? b1b : b1a;
    const float* W2 = branch ? W2b : W2a;
    const float* b2 = branch ? b2b : b2a;

    __shared__ float xs[ROWS_PER_WG + Tw];      /* 576 floats of x slice    */
    __shared__ float w1s[NK1 * Tw];             /* 1024                      */
    __shared__ float w2s[NK2 * NK1];            /* 128                       */
    __shared__ float b1s[NK1];
    __shared__ float b2s[NK2];
    __shared__ float d1s[8][16][17];            /* per-wave d1 tile, padded  */

    const int tid = threadIdx.x;

    /* ---- stage inputs into LDS ---------------------------------------- */
    for (int i = tid; i < ROWS_PER_WG + Tw; i += 256) {
        int gi = l0g + i;
        if (gi > Lfull - 1) gi = Lfull - 1;     /* clamp: values unused OOB */
        xs[i] = x[(size_t)b * Lfull + gi];
    }
    for (int i = tid; i < NK1 * Tw; i += 256) w1s[i] = W1[i];
    for (int i = tid; i < NK2 * NK1; i += 256) w2s[i] = W2[i];
    if (tid < NK1) b1s[tid] = b1[tid];
    if (tid < NK2) b2s[tid] = b2[tid];
    __syncthreads();

    const int wave = tid >> 5;
    const int lane = tid & 31;
    const int half = lane >> 4;         /* lane group 0..15 vs 16..31 */
    const int mn   = lane & 15;         /* A-row M / B,C column N      */

    /* ---- pack constant B operands (W1) once per wave ------------------- */
    /* B 32x16 (f16): half-index i -> K = i + (i>=8 ? 8:0) + 8*half, column N = mn */
    v16h bm0, bm1;
#pragma unroll
    for (int i = 0; i < 16; ++i) {
        const int kb = i + ((i & 8) ? 8 : 0) + 8 * half;
        bm0[i] = (_Float16)w1s[mn * Tw + kb];        /* B[t,k1] = W1[k1,t] */
        bm1[i] = (_Float16)w1s[mn * Tw + 32 + kb];
    }
    const float bias1 = b1s[mn];

    /* ---- 4 tiles per wave, reusing the packed B fragments -------------- */
#pragma unroll
    for (int tt = 0; tt < TILES_PER_WAVE; ++tt) {
        const int rowbase = wave * 16 + tt * 128;   /* tile base within WG */

        /* A 16x32 (f16): half-index i -> K = i + (i>=8 ? 8:0) + 8*half    */
        v16h a0, a1;
#pragma unroll
        for (int i = 0; i < 16; ++i) {
            const int ka = i + ((i & 8) ? 8 : 0) + 8 * half;
            a0[i] = (_Float16)xs[rowbase + mn + ka];
            a1[i] = (_Float16)xs[rowbase + mn + 32 + ka];
        }

        /* d1 tile = relu(H x W1^T + b1) via two accumulating WMMAs */
        v8f c = {};
        c = __builtin_amdgcn_wmma_f32_16x16x32_f16(false, a0, false, bm0,
                                                   (short)0, c, false, false);
        c = __builtin_amdgcn_wmma_f32_16x16x32_f16(false, a1, false, bm1,
                                                   (short)0, c, false, false);

#pragma unroll
        for (int r = 0; r < 8; ++r) {            /* C: M = r + 8*half, N = mn */
            float v = c[r] + bias1;
            d1s[wave][r + 8 * half][mn] = v > 0.0f ? v : 0.0f;
        }
        __syncthreads();

        /* second layer (K1=16 -> K2=8), argmax/max, outputs */
        if (lane < 16) {
            const int lp = l0g + rowbase + lane; /* global l' index */
            if (lp < LP) {
                float d2v[NK2];
                float mx = -1.0e30f; int mi = 0;
#pragma unroll
                for (int j = 0; j < NK2; ++j) {
                    float acc = b2s[j];
#pragma unroll
                    for (int k = 0; k < NK1; ++k)
                        acc = fmaf(d1s[wave][lane][k], w2s[j * NK1 + k], acc);
                    acc = acc > 0.0f ? acc : 0.0f;
                    d2v[j] = acc;
                    if (acc > mx) { mx = acc; mi = j; } /* '>' keeps 1st max */
                }
                const size_t row  = (size_t)b * LP + lp;
                const size_t base = (size_t)branch * BR_STRIDE;

                float4* dd = (float4*)(out + base + row * NK2); /* 32B aligned */
                dd[0] = make_float4(d2v[0], d2v[1], d2v[2], d2v[3]);
                dd[1] = make_float4(d2v[4], d2v[5], d2v[6], d2v[7]);

                out[base + SZD + row]       = (float)mi;  /* hard_idx        */
                out[base + SZD + SZI + row] = mx;         /* vals == soma_in */
                if (useT)                                  /* coalesced LIF feed */
                    IT[(size_t)branch * LP * Bsz + (size_t)lp * Bsz + b] = mx;
            }
        }
        __syncthreads();    /* WAR: next tile rewrites d1s */
    }
}

/* ---- LIF membrane scan: 512 independent chains, strictly sequential ----
 * Reset non-linearity forbids a parallel scan; instead keep the load stream
 * off the critical path: 8 independent loads batched per chunk + prefetch. */
__global__ __launch_bounds__(256) void lif_scan_kernel(
    float* __restrict__ out, const float* __restrict__ IT, int useT)
{
    const int branch = blockIdx.x;
    const int b      = threadIdx.x;              /* B == 256 == blockDim.x */
    const size_t base = (size_t)branch * BR_STRIDE;
    const float* vals = out + base + SZD + SZI;
    const float* It   = IT + (size_t)branch * LP * Bsz;

    const float decay = 1.0f - 1.0f / TAUf;      /* v += -v/TAU + I */
    float v = 0.0f, ssum = 0.0f, wsum = 0.0f;
    int first = -1;

#pragma unroll 1
    for (int t0 = 0; t0 < LP - 1; t0 += 8) {     /* 4032 = 8 * 504 */
        if (t0 + 64 < LP) {                      /* pull stream 64 steps ahead */
            const float* pf = useT ? &It[(size_t)(t0 + 64) * Bsz + b]
                                   : &vals[(size_t)b * LP + (t0 + 64)];
            __builtin_prefetch(pf, 0, 1);        /* -> global_prefetch_b8 */
        }
        float iv[8];
#pragma unroll
        for (int u = 0; u < 8; ++u)              /* 8 independent loads */
            iv[u] = useT ? It[(size_t)(t0 + u) * Bsz + b]
                         : vals[(size_t)b * LP + (t0 + u)];
#pragma unroll
        for (int u = 0; u < 8; ++u) {            /* dependent membrane chain */
            v = fmaf(v, decay, iv[u]);
            if (v >= THf) {                      /* forward s = s_hard */
                if (first < 0) first = t0 + u;
                ssum += 1.0f;
                wsum += (float)(t0 + u);
                v = 0.0f;                        /* hard reset */
            }
        }
    }
    {   /* remainder step t = LP-1 */
        const int t = LP - 1;
        const float i_t = useT ? It[(size_t)t * Bsz + b]
                               : vals[(size_t)b * LP + t];
        v = fmaf(v, decay, i_t);
        if (v >= THf) {
            if (first < 0) first = t;
            ssum += 1.0f;
            wsum += (float)t;
        }
    }
    out[base + SZD + 2 * SZI + b]       = (float)(first < 0 ? 0 : first);
    out[base + SZD + 2 * SZI + Bsz + b] = wsum / (ssum + 1e-6f);
}

extern "C" void kernel_launch(void* const* d_in, const int* in_sizes, int n_in,
                              void* d_out, int out_size, void* d_ws, size_t ws_size,
                              hipStream_t stream)
{
    (void)in_sizes; (void)n_in; (void)out_size;
    const float* xa  = (const float*)d_in[0];
    const float* xb  = (const float*)d_in[1];
    const float* W1a = (const float*)d_in[2];
    const float* b1a = (const float*)d_in[3];
    const float* W2a = (const float*)d_in[4];
    const float* b2a = (const float*)d_in[5];
    const float* W1b = (const float*)d_in[6];
    const float* b1b = (const float*)d_in[7];
    const float* W2b = (const float*)d_in[8];
    const float* b2b = (const float*)d_in[9];
    float* out = (float*)d_out;

    const size_t needT = 2 * (size_t)LP * Bsz * sizeof(float);
    const int useT = (d_ws != nullptr && ws_size >= needT) ? 1 : 0;
    float* IT = useT ? (float*)d_ws : out;       /* dummy ptr if unused */

    dim3 grid((LP + ROWS_PER_WG - 1) / ROWS_PER_WG, Bsz, 2);  /* 8 x 256 x 2 */
    dendrite_wmma_kernel<<<grid, 256, 0, stream>>>(
        xa, xb, W1a, b1a, W2a, b2a, W1b, b1b, W2b, b2b, out, IT, useT);

    lif_scan_kernel<<<dim3(2), 256, 0, stream>>>(out, IT, useT);
}
